// Block_15032385536608
// MI455X (gfx1250) — matrix operand order repaired and checked by reading, not compiled, not run on hardware
//
#include <hip/hip_runtime.h>
#include <hip/hip_bf16.h>
#include <math.h>

// MI455X / gfx1250, wave32. All GEMMs run on v_wmma_f32_16x16x32_bf16.
// B operands pre-transposed to [N][K] bf16 so B-fragments are two contiguous 16B loads per lane.
// GEMM: 32x64 output tile per wave (2 A-frags x 4 B-frags -> 8 WMMAs per k-chunk)
// for 4x A-reuse / 2x B-reuse vs a 16x16 tile; global_prefetch_b8 one panel ahead.

typedef __bf16 bf16;
typedef __attribute__((ext_vector_type(8)))  bf16  v8bf;
typedef __attribute__((ext_vector_type(16))) bf16  v16bf;
typedef __attribute__((ext_vector_type(8)))  float v8f;

// A-fragment (16x32 bf16, M x K): lane = h*16 + r holds row M=r.
// elements e<8 -> K = kbase + h*8 + e ; e>=8 -> K = kbase + 16 + h*8 + (e-8).
static __device__ __forceinline__ v16bf load_a_frag(const bf16* __restrict__ A, int lda,
                                                    int m0, int kbase, int h, int r) {
  const bf16* p = A + (size_t)(m0 + r) * lda + kbase + h * 8;
  v8bf lo = *(const v8bf*)p;
  v8bf hi = *(const v8bf*)(p + 16);
  v16bf a;
#pragma unroll
  for (int i = 0; i < 8; ++i) { a[i] = lo[i]; a[i + 8] = hi[i]; }
  return a;
}

// B-fragment (32x16 bf16, K x N) loaded from BT = [N][K] row-major:
// lane = h*16 + n holds col N=n, elements e<8 -> K = kbase + h*8 + e ; e>=8 -> K = kbase + 16 + h*8 + (e-8).
static __device__ __forceinline__ v16bf load_b_frag(const bf16* __restrict__ BT, int ldb,
                                                    int n0, int kbase, int h, int n) {
  const bf16* p = BT + (size_t)(n0 + n) * ldb + kbase + h * 8;
  v8bf lo = *(const v8bf*)p;
  v8bf hi = *(const v8bf*)(p + 16);
  v16bf b;
#pragma unroll
  for (int i = 0; i < 8; ++i) { b[i] = lo[i]; b[i + 8] = hi[i]; }
  return b;
}

// ---------------- 32x64-tile-per-wave bf16 GEMM ----------------
// D = A[MxK] * BT^T[KxN] (+bias) (+relu) (+resid) ; store f32 / bf16 / bf16-transposed
__global__ void gemm_kernel(const bf16* __restrict__ A, const bf16* __restrict__ BT,
                            const float* __restrict__ bias, const float* __restrict__ resid,
                            void* __restrict__ out, int M, int N, int K,
                            int store_mode, int do_relu, int ldo)
{
  const int wid  = threadIdx.x >> 5;
  const int wave = blockIdx.x * (blockDim.x >> 5) + wid;
  const int tn_cnt = N >> 6;                 // 64-wide N tiles
  const int total  = (M >> 5) * tn_cnt;      // 32-high M tiles
  if (wave >= total) return;
  const int m0 = (wave / tn_cnt) << 5;
  const int n0 = (wave % tn_cnt) << 6;
  const int lane = threadIdx.x & 31;
  const int h = lane >> 4, ln = lane & 15;

  v8f acc[2][4];
  v8f zero = {};
#pragma unroll
  for (int i = 0; i < 2; ++i)
#pragma unroll
    for (int j = 0; j < 4; ++j) acc[i][j] = zero;

  for (int kb = 0; kb < K; kb += 32) {
    // stream prefetch one 256-element (512B) panel ahead (global_prefetch_b8)
    if (kb + 256 < K) {
      __builtin_prefetch(A  + (size_t)(m0 + lane) * K + kb + 256, 0, 1);
      __builtin_prefetch(BT + (size_t)(n0 + lane) * K + kb + 256, 0, 1);
      __builtin_prefetch(BT + (size_t)(n0 + 32 + lane) * K + kb + 256, 0, 1);
    }
    v16bf a0 = load_a_frag(A, K, m0,      kb, h, ln);
    v16bf a1 = load_a_frag(A, K, m0 + 16, kb, h, ln);
#pragma unroll
    for (int j = 0; j < 4; ++j) {
      v16bf b = load_b_frag(BT, K, n0 + j * 16, kb, h, ln);
      acc[0][j] = __builtin_amdgcn_wmma_f32_16x16x32_bf16(false, a0, false, b, (short)0,
                                                          acc[0][j], false, false);
      acc[1][j] = __builtin_amdgcn_wmma_f32_16x16x32_bf16(false, a1, false, b, (short)0,
                                                          acc[1][j], false, false);
    }
  }

  // D layout: lane (h*16+ln), VGPR r -> (M = mtile + r + 8h, N = ntile + ln)
#pragma unroll
  for (int i = 0; i < 2; ++i) {
#pragma unroll
    for (int j = 0; j < 4; ++j) {
#pragma unroll
      for (int r = 0; r < 8; ++r) {
        const int m = m0 + i * 16 + r + 8 * h;
        const int n = n0 + j * 16 + ln;
        float v = acc[i][j][r];
        if (bias)    v += bias[n];
        if (do_relu) v = v > 0.f ? v : 0.f;
        if (resid)   v += resid[(size_t)m * ldo + n];
        if (store_mode == 0)      ((float*)out)[(size_t)m * ldo + n] = v;
        else if (store_mode == 1) ((bf16*)out)[(size_t)m * ldo + n] = (bf16)v;
        else                      ((bf16*)out)[(size_t)n * ldo + m] = (bf16)v; // transposed
      }
    }
  }
}

// ---------------- wave-level flash attention ----------------
// One wave owns a 16-row Q tile of one (batch, head). Streams causal S in
// 32-wide blocks: QK^T via 4 WMMAs, online softmax with 16-lane shfl
// reductions, P re-striped D-layout -> A-layout through LDS, PV via 4 WMMAs.
__global__ void attn_kernel(const bf16* __restrict__ q, const bf16* __restrict__ k,
                            const bf16* __restrict__ vT, bf16* __restrict__ ctx,
                            int T, int NH, int HD)
{
  __shared__ bf16 pbuf[8][16][32];   // per-wave P tile (16 x 32 bf16)
  const int wid  = threadIdx.x >> 5;
  const int wave = blockIdx.x * 8 + wid;
  const int qtiles = T >> 4;
  const int bb   = wave / (NH * qtiles);
  const int rem  = wave % (NH * qtiles);
  const int head = rem / qtiles;
  const int q0   = (rem % qtiles) << 4;
  const int lane = threadIdx.x & 31;
  const int h = lane >> 4, ln = lane & 15;
  const int CC = NH * HD;

  const bf16* qb = q  + (size_t)bb * T * CC + head * HD;      // [T, CC] rows
  const bf16* kb = k  + (size_t)bb * T * CC + head * HD;      // [T, CC] rows (= (K^T)^T)
  const bf16* vb = vT + (size_t)(bb * NH + head) * HD * T;    // [HD, T]  (= (V)^T)

  const v16bf aq0 = load_a_frag(qb, CC, q0, 0,  h, ln);       // Q rows, K-chunk 0..31
  const v16bf aq1 = load_a_frag(qb, CC, q0, 32, h, ln);       // K-chunk 32..63

  float mi[8], li[8];
  v8f acc[4];
  v8f zero = {};
#pragma unroll
  for (int r = 0; r < 8; ++r) { mi[r] = -3.0e38f; li[r] = 0.f; }
#pragma unroll
  for (int j = 0; j < 4; ++j) acc[j] = zero;

  const float scale = 0.125f;            // 1/sqrt(HD=64)
  const int send = q0 + 16;              // causal bound (exclusive)

  for (int sb = 0; sb < send; sb += 32) {
    v8f s0 = {}, s1 = {};
    {
      v16bf b0 = load_b_frag(kb, CC, sb,      0,  h, ln);
      s0 = __builtin_amdgcn_wmma_f32_16x16x32_bf16(false, aq0, false, b0, (short)0, s0, false, false);
      v16bf b1 = load_b_frag(kb, CC, sb,      32, h, ln);
      s0 = __builtin_amdgcn_wmma_f32_16x16x32_bf16(false, aq1, false, b1, (short)0, s0, false, false);
      v16bf b2 = load_b_frag(kb, CC, sb + 16, 0,  h, ln);
      s1 = __builtin_amdgcn_wmma_f32_16x16x32_bf16(false, aq0, false, b2, (short)0, s1, false, false);
      v16bf b3 = load_b_frag(kb, CC, sb + 16, 32, h, ln);
      s1 = __builtin_amdgcn_wmma_f32_16x16x32_bf16(false, aq1, false, b3, (short)0, s1, false, false);
    }

#pragma unroll
    for (int r = 0; r < 8; ++r) {
      const int mg = q0 + r + 8 * h;                       // this lane's global q row
      float v0 = (sb + ln      <= mg) ? s0[r] * scale : -3.0e38f;
      float v1 = (sb + 16 + ln <= mg) ? s1[r] * scale : -3.0e38f;
      float mx = fmaxf(v0, v1);
#pragma unroll
      for (int msk = 1; msk < 16; msk <<= 1)               // stays within 16-lane half
        mx = fmaxf(mx, __shfl_xor(mx, msk, 32));
      const float mnew = fmaxf(mi[r], mx);
      const float fac  = __expf(mi[r] - mnew);
      mi[r] = mnew;
      const float p0 = __expf(v0 - mnew);
      const float p1 = __expf(v1 - mnew);
      float rs = p0 + p1;
#pragma unroll
      for (int msk = 1; msk < 16; msk <<= 1)
        rs += __shfl_xor(rs, msk, 32);
      li[r] = li[r] * fac + rs;
#pragma unroll
      for (int j = 0; j < 4; ++j) acc[j][r] *= fac;
      pbuf[wid][r + 8 * h][ln]      = (bf16)p0;            // D-layout -> row-major LDS
      pbuf[wid][r + 8 * h][ln + 16] = (bf16)p1;
    }

    // same-wave LDS RAW: wait for DS stores before re-striping
    asm volatile("s_wait_dscnt 0" ::: "memory");

    v16bf ap;                                              // P as A-fragment (row = ln)
#pragma unroll
    for (int i = 0; i < 8; ++i) {
      ap[i]     = pbuf[wid][ln][h * 8 + i];
      ap[i + 8] = pbuf[wid][ln][16 + h * 8 + i];
    }

#pragma unroll
    for (int j = 0; j < 4; ++j) {                          // ctx cols j*16..j*16+15
      v16bf bv = load_b_frag(vb, T, j * 16, sb, h, ln);
      acc[j] = __builtin_amdgcn_wmma_f32_16x16x32_bf16(false, ap, false, bv, (short)0,
                                                       acc[j], false, false);
    }
  }

#pragma unroll
  for (int r = 0; r < 8; ++r) {
    const float inv = 1.0f / li[r];
    const int m = q0 + r + 8 * h;
#pragma unroll
    for (int j = 0; j < 4; ++j)
      ctx[((size_t)bb * T + m) * CC + head * HD + j * 16 + ln] = (bf16)(acc[j][r] * inv);
  }
}

// ---------------- LayerNorm (torch-style: unbiased std, eps on std) -> bf16 ----------------
__global__ void ln_kernel(const float* __restrict__ x, const float* __restrict__ g,
                          const float* __restrict__ be, bf16* __restrict__ out, int C)
{
  __shared__ float red[256];
  const int row = blockIdx.x;
  const int tid = threadIdx.x;
  const float* xr = x + (size_t)row * C;

  float s = 0.f;
  for (int c = tid; c < C; c += blockDim.x) s += xr[c];
  red[tid] = s; __syncthreads();
  for (int off = 128; off > 0; off >>= 1) {
    if (tid < off) red[tid] += red[tid + off];
    __syncthreads();
  }
  const float mean = red[0] / C;
  __syncthreads();

  float s2 = 0.f;
  for (int c = tid; c < C; c += blockDim.x) { float d = xr[c] - mean; s2 += d * d; }
  red[tid] = s2; __syncthreads();
  for (int off = 128; off > 0; off >>= 1) {
    if (tid < off) red[tid] += red[tid + off];
    __syncthreads();
  }
  const float stdv = sqrtf(red[0] / (C - 1));
  const float inv  = 1.0f / (stdv + 1e-6f);

  for (int c = tid; c < C; c += blockDim.x)
    out[(size_t)row * C + c] = (bf16)(g[c] * (xr[c] - mean) * inv + be[c]);
}

// ---------------- fp32 -> bf16 convert + per-group transpose ----------------
// in: [G][R][Cc] f32  ->  out: [G][Cc][R] bf16   (gives BT = [N][K] layouts)
__global__ void convT_kernel(const float* __restrict__ in, bf16* __restrict__ out,
                             int G, int R, int Cc)
{
  const size_t i = (size_t)blockIdx.x * blockDim.x + threadIdx.x;
  const size_t total = (size_t)G * R * Cc;
  if (i >= total) return;
  const size_t gg  = i / ((size_t)R * Cc);
  const size_t rem = i % ((size_t)R * Cc);
  const size_t r = rem / Cc, c = rem % Cc;
  out[gg * (size_t)R * Cc + c * (size_t)R + r] = (bf16)in[i];
}

extern "C" void kernel_launch(void* const* d_in, const int* in_sizes, int n_in,
                              void* d_out, int out_size, void* d_ws, size_t ws_size,
                              hipStream_t stream)
{
  (void)in_sizes; (void)n_in; (void)out_size; (void)ws_size;
  const int B = 2, T = 2048, C = 1024, NH = 16, HD = 64, FF = 4;
  const int M  = B * T;      // 4096 token rows
  const int FC = FF * C;     // 4096

  const float* x   = (const float*)d_in[0];
  const float* Wq  = (const float*)d_in[1];
  const float* Wk  = (const float*)d_in[2];
  const float* Wv  = (const float*)d_in[3];
  const float* Wo  = (const float*)d_in[4];
  const float* bo  = (const float*)d_in[5];
  const float* W1  = (const float*)d_in[6];
  const float* b1  = (const float*)d_in[7];
  const float* W2  = (const float*)d_in[8];
  const float* b2  = (const float*)d_in[9];
  const float* g1  = (const float*)d_in[10];
  const float* be1 = (const float*)d_in[11];
  const float* g2  = (const float*)d_in[12];
  const float* be2 = (const float*)d_in[13];
  float* out = (float*)d_out;

  size_t off = 0;
  auto wsa = [&](size_t bytes) -> void* {
    void* p = (char*)d_ws + off;
    off += (bytes + 255) & ~(size_t)255;
    return p;
  };

  bf16* xn   = (bf16*)wsa((size_t)M * C * 2);   // LN1(x)
  bf16* WqT  = (bf16*)wsa((size_t)C * C * 2);   // [NH*HD][C]
  bf16* WkT  = (bf16*)wsa((size_t)C * C * 2);
  bf16* WvT  = (bf16*)wsa((size_t)C * C * 2);
  bf16* WoT  = (bf16*)wsa((size_t)C * C * 2);   // [C][NH*HD]
  bf16* W1T  = (bf16*)wsa((size_t)C * FC * 2);  // [FC][C]
  bf16* W2T  = (bf16*)wsa((size_t)FC * C * 2);  // [C][FC]
  bf16* qb   = (bf16*)wsa((size_t)M * C * 2);   // [B,T,NH,HD]
  bf16* kbuf = (bf16*)wsa((size_t)M * C * 2);   // [B,T,NH,HD]
  bf16* vT   = (bf16*)wsa((size_t)M * C * 2);   // [B,NH,HD,T]
  bf16* ctx  = (bf16*)wsa((size_t)M * C * 2);   // [B,T,NH*HD]
  float* x1  = (float*)wsa((size_t)M * C * 4);  // x + attn_out (residual)
  bf16* xn2  = (bf16*)wsa((size_t)M * C * 2);   // LN2(x1)
  bf16* ffh  = (bf16*)wsa((size_t)M * FC * 2);  // relu(xn2 @ W1 + b1)

  const int blk = 256;
  {
    int tot = NH * C * HD;
    convT_kernel<<<(tot + blk - 1) / blk, blk, 0, stream>>>(Wq, WqT, NH, C, HD);
    convT_kernel<<<(tot + blk - 1) / blk, blk, 0, stream>>>(Wk, WkT, NH, C, HD);
    convT_kernel<<<(tot + blk - 1) / blk, blk, 0, stream>>>(Wv, WvT, NH, C, HD);
    tot = C * C;
    convT_kernel<<<(tot + blk - 1) / blk, blk, 0, stream>>>(Wo, WoT, 1, C, C);
    tot = C * FC;
    convT_kernel<<<(tot + blk - 1) / blk, blk, 0, stream>>>(W1, W1T, 1, C, FC);
    convT_kernel<<<(tot + blk - 1) / blk, blk, 0, stream>>>(W2, W2T, 1, FC, C);
  }

  ln_kernel<<<M, blk, 0, stream>>>(x, g1, be1, xn, C);

  auto gemm = [&](const bf16* A, const bf16* BT, const float* bias, const float* resid,
                  void* o, int mm, int nn, int kk, int mode, int relu, int ldo) {
    int waves  = (mm >> 5) * (nn >> 6);   // 32x64 tile per wave
    int blocks = (waves + 7) / 8;         // 8 waves per 256-thread block
    gemm_kernel<<<blocks, blk, 0, stream>>>(A, BT, bias, resid, o, mm, nn, kk,
                                            mode, relu, ldo);
  };

  // Q, K row-major; V written transposed per batch for contiguous PV B-fragments
  gemm(xn, WqT, nullptr, nullptr, qb,   M, C, C, /*bf16*/1, 0, C);
  gemm(xn, WkT, nullptr, nullptr, kbuf, M, C, C, /*bf16*/1, 0, C);
  for (int b = 0; b < B; ++b)
    gemm(xn + (size_t)b * T * C, WvT, nullptr, nullptr,
         vT + (size_t)b * C * T, T, C, C, /*bf16^T*/2, 0, T);

  attn_kernel<<<(B * NH * (T >> 4)) / 8, blk, 0, stream>>>(qb, kbuf, vT, ctx, T, NH, HD);

  gemm(ctx, WoT, bo, x,  x1,  M, C,  C,  /*f32*/0, 0, C);    // x1 = x + ctx@Wo + bo
  ln_kernel<<<M, blk, 0, stream>>>(x1, g2, be2, xn2, C);
  gemm(xn2, W1T, b1, nullptr, ffh, M, FC, C,  /*bf16*/1, 1, FC);  // relu
  gemm(ffh, W2T, b2, x1, out, M, C,  FC, /*f32*/0, 0, C);    // out = x1 + ffh@W2 + b2
}
